// Decoder_66494683676994
// MI455X (gfx1250) — matrix operand order repaired and checked by reading, not compiled, not run on hardware
//
#include <hip/hip_runtime.h>

typedef __attribute__((ext_vector_type(16))) __bf16 v16bf;
typedef __attribute__((ext_vector_type(8)))  __bf16 v8bf;
typedef __attribute__((ext_vector_type(8)))  float  v8f;

static __device__ __forceinline__ __bf16 f2bf(float f) {
  unsigned int u = __builtin_bit_cast(unsigned int, f);
  u += 0x7FFFu + ((u >> 16) & 1u);                  // round-to-nearest-even
  return __builtin_bit_cast(__bf16, (unsigned short)(u >> 16));
}
static __device__ __forceinline__ __bf16 bf0() {
  return __builtin_bit_cast(__bf16, (unsigned short)0);
}

// Convert x (fp32, [C][S] channel-major) -> xb (bf16, [S][C] channel-last),
// and zero the 128-element zero-pad block used for conv padding reads.
__global__ void cvt_x_kernel(const float* __restrict__ x, __bf16* __restrict__ xb,
                             __bf16* __restrict__ zbuf, int C, int S) {
  int t = blockIdx.x * blockDim.x + threadIdx.x;
  if (t < 128) zbuf[t] = bf0();
  if (t < C * S) {
    int c = t % C, sp = t / C;
    xb[t] = f2bf(x[c * S + sp]);
  }
}

// One-time weight repack into WMMA A-fragment order (bf16, zero-padded rows):
// pk[((mt*27 + tap)*(CIN/32) + chunk)*512 + laneS*16 + j]
//   = w[co=mt*16+(laneS&15)][ci=chunk*32+aoff(j,laneS>>4)][tap]
__global__ void pack_w_kernel(const float* __restrict__ w, __bf16* __restrict__ pk,
                              int CIN, int COUT, int MT) {
  int t = blockIdx.x * blockDim.x + threadIdx.x;
  int total = MT * 27 * (CIN / 32) * 512;
  if (t >= total) return;
  int j     = t & 15;
  int laneS = (t >> 4) & 31;
  int chunk = (t >> 9) % (CIN / 32);
  int rest  = (t >> 9) / (CIN / 32);
  int tap   = rest % 27;
  int mt    = rest / 27;
  int i = j >> 1, bs = j & 1;
  // A 16x32 bf16 layout (contiguous, matches B): lanes 0-15 K{0..15},
  // lanes 16-31 K{16..31}; row = lane&15.
  int aoff = 2 * i + bs + 16 * (laneS >> 4);
  int ci = chunk * 32 + aoff;
  int co = mt * 16 + (laneS & 15);
  pk[t] = (co < COUT) ? f2bf(w[(size_t)(co * CIN + ci) * 27 + tap]) : bf0();
}

// One wave: 16(out-ch) x 16(spatial) tile of 3x3x3 conv (UP=0) or stride-2
// transposed conv (UP=1), implicit GEMM, K = 27*CIN via v_wmma_f32_16x16x32_bf16.
// B: bf16 channel-last (contiguous 32B fragment loads, padding via zbuf redirect).
// A: pre-packed fragment-order bf16, bulk-copied per tap into LDS (4KB).
template <int CIN, int COUT, int DIN, int DOUT, int UP, int RELU, int WF, int WB>
__global__ __launch_bounds__(256) void dec_gemm(
    const __bf16* __restrict__ bin, const __bf16* __restrict__ packA,
    const float* __restrict__ bias, float* __restrict__ outF,
    __bf16* __restrict__ outB, const __bf16* __restrict__ zbuf)
{
  constexpr int DOUT3 = DOUT * DOUT * DOUT;
  constexpr int ATILE = CIN * 16;                 // bf16 elems per (Mtile, tap)
  __shared__ alignas(32) __bf16 ldsA[ATILE];

  const int lane = threadIdx.x & 31;
  const int wave = threadIdx.x >> 5;
  const int half = lane >> 4;
  const int l16  = lane & 15;
  const int n    = (blockIdx.x * 8 + wave) * 16 + l16;  // output voxel (column)
  const int coB  = blockIdx.y * 16;                     // out-channel tile base

  const int ox = n % DOUT;
  const int oy = (n / DOUT) % DOUT;
  const int oz = n / (DOUT * DOUT);

  const __bf16* aSrc = packA + (size_t)blockIdx.y * (27 * ATILE);
  const int eCopy = threadIdx.x * 8;              // 16B per thread per tap

  v8f c;
  #pragma unroll
  for (int r = 0; r < 8; ++r) {                   // D: VGPR r = row coB+r+8*half
    int co = coB + r + 8 * half;
    bool inr = (COUT % 16 == 0) || (co < COUT);
    c[r] = inr ? bias[inr ? co : 0] : 0.0f;
  }

  int kx = 0, ky = 0, kz = 0;                     // incremental tap decomposition
  #pragma unroll 1
  for (int tap = 0; tap < 27; ++tap) {
    __syncthreads();                              // previous tap's A reads done
    if (eCopy < ATILE)                            // contiguous 4KB tile -> LDS
      *(v8bf*)(ldsA + eCopy) = *(const v8bf*)(aSrc + tap * ATILE + eCopy);
    __syncthreads();

    int iz, iy, ix; bool valid;
    if (UP) {                                     // lhs-dilation 2: parity + range
      int dz = oz + kz - 1, dy = oy + ky - 1, dx = ox + kx - 1;
      valid = (((dz | dy | dx) & 1) == 0);
      iz = dz >> 1; iy = dy >> 1; ix = dx >> 1;
    } else {                                      // plain conv, pad 1
      iz = oz + kz - 1; iy = oy + ky - 1; ix = ox + kx - 1;
      valid = true;
    }
    valid = valid && (unsigned)iz < (unsigned)DIN && (unsigned)iy < (unsigned)DIN &&
            (unsigned)ix < (unsigned)DIN;
    const int sp = (iz * DIN + iy) * DIN + ix;
    // Padding -> redirect the whole fragment load to a zeroed scratch block:
    const __bf16* bsrc = valid ? (bin + (size_t)sp * CIN) : zbuf;

    #pragma unroll
    for (int ch = 0; ch < CIN / 32; ++ch) {
      v16bf a  = *(const v16bf*)(ldsA + ((ch * 32 + lane) << 4));     // 32B ds read
      v16bf bm = *(const v16bf*)(bsrc + ch * 32 + 16 * half);         // 32B vmem read
      c = __builtin_amdgcn_wmma_f32_16x16x32_bf16(
            false, a, false, bm, (short)0, c, false, false);
    }

    if (++kx == 3) { kx = 0; if (++ky == 3) { ky = 0; ++kz; } }
  }

  if (WB) {                                       // bf16 channel-last (next stage in)
    v8bf pkv;
    #pragma unroll
    for (int r = 0; r < 8; ++r)
      pkv[r] = f2bf(RELU ? fmaxf(c[r], 0.0f) : c[r]);
    *(v8bf*)(outB + (size_t)n * COUT + coB + 8 * half) = pkv;
  }
  if (WF) {                                       // fp32 channel-major (d_out)
    #pragma unroll
    for (int r = 0; r < 8; ++r) {
      int co = coB + r + 8 * half;
      if ((COUT % 16 == 0) || (co < COUT)) {
        float v = c[r];
        if (RELU) v = fmaxf(v, 0.0f);
        outF[(size_t)co * DOUT3 + n] = v;
      }
    }
  }
}

extern "C" void kernel_launch(void* const* d_in, const int* in_sizes, int n_in,
                              void* d_out, int out_size, void* d_ws, size_t ws_size,
                              hipStream_t stream) {
  (void)in_sizes; (void)n_in; (void)out_size;
  const float* x       = (const float*)d_in[0];
  const float* w_up0   = (const float*)d_in[1];
  const float* b_up0   = (const float*)d_in[2];
  const float* w_conv0 = (const float*)d_in[3];
  const float* b_conv0 = (const float*)d_in[4];
  const float* w_up1   = (const float*)d_in[5];
  const float* b_up1   = (const float*)d_in[6];
  const float* w_conv1 = (const float*)d_in[7];
  const float* b_conv1 = (const float*)d_in[8];
  const float* w_up2   = (const float*)d_in[9];
  const float* b_up2   = (const float*)d_in[10];
  const float* w_conv2 = (const float*)d_in[11];
  const float* b_conv2 = (const float*)d_in[12];

  // d_out = out2 (128x16^3) | out1 (128x32^3) | out0 (3x64^3), fp32 channel-major
  float* out2 = (float*)d_out;
  float* out1 = out2 + (size_t)128 * 4096;
  float* out0 = out1 + (size_t)128 * 32768;

  // ws: zero pad | bf16 channel-last intermediates | packed bf16 weights
  char* ws = (char*)d_ws;
  size_t off = 0;
  __bf16* zbuf = (__bf16*)(ws + off); off += 256;                    // 128 bf16
  __bf16* xb   = (__bf16*)(ws + off); off += (size_t)128 * 512 * 2;
  __bf16* t0b  = (__bf16*)(ws + off); off += (size_t)128 * 4096 * 2;
  __bf16* o2b  = (__bf16*)(ws + off); off += (size_t)128 * 4096 * 2;
  __bf16* t1b  = (__bf16*)(ws + off); off += (size_t)128 * 32768 * 2;
  __bf16* o1b  = (__bf16*)(ws + off); off += (size_t)128 * 32768 * 2;
  __bf16* t2b  = (__bf16*)(ws + off); off += (size_t)64 * 262144 * 2;
  const size_t pk8_128 = (size_t)8 * 27 * 4 * 512;   // MT=8, CIN=128 (elems)
  const size_t pk4_128 = (size_t)4 * 27 * 4 * 512;   // MT=4, CIN=128
  const size_t pk1_64  = (size_t)1 * 27 * 2 * 512;   // MT=1, CIN=64
  __bf16* pU0 = (__bf16*)(ws + off); off += pk8_128 * 2;
  __bf16* pC0 = (__bf16*)(ws + off); off += pk8_128 * 2;
  __bf16* pU1 = (__bf16*)(ws + off); off += pk8_128 * 2;
  __bf16* pC1 = (__bf16*)(ws + off); off += pk8_128 * 2;
  __bf16* pU2 = (__bf16*)(ws + off); off += pk4_128 * 2;
  __bf16* pC2 = (__bf16*)(ws + off); off += pk1_64 * 2;
  if (ws_size < off) return;

  dim3 blk(256);
  cvt_x_kernel<<<dim3(256), blk, 0, stream>>>(x, xb, zbuf, 128, 512);
  pack_w_kernel<<<dim3((int)(pk8_128 / 256)), blk, 0, stream>>>(w_up0,   pU0, 128, 128, 8);
  pack_w_kernel<<<dim3((int)(pk8_128 / 256)), blk, 0, stream>>>(w_conv0, pC0, 128, 128, 8);
  pack_w_kernel<<<dim3((int)(pk8_128 / 256)), blk, 0, stream>>>(w_up1,   pU1, 128, 128, 8);
  pack_w_kernel<<<dim3((int)(pk8_128 / 256)), blk, 0, stream>>>(w_conv1, pC1, 128, 128, 8);
  pack_w_kernel<<<dim3((int)(pk4_128 / 256)), blk, 0, stream>>>(w_up2,   pU2, 128,  64, 4);
  pack_w_kernel<<<dim3((int)(pk1_64  / 256)), blk, 0, stream>>>(w_conv2, pC2,  64,   3, 1);

  // grid.x = DOUT^3 / (16 cols * 8 waves), grid.y = COUT tiles
  dec_gemm<128, 128,  8, 16, 1, 0, 0, 1><<<dim3(  32, 8), blk, 0, stream>>>(
      xb,  pU0, b_up0,   nullptr, t0b, zbuf);
  dec_gemm<128, 128, 16, 16, 0, 1, 1, 1><<<dim3(  32, 8), blk, 0, stream>>>(
      t0b, pC0, b_conv0, out2,    o2b, zbuf);
  dec_gemm<128, 128, 16, 32, 1, 0, 0, 1><<<dim3( 256, 8), blk, 0, stream>>>(
      o2b, pU1, b_up1,   nullptr, t1b, zbuf);
  dec_gemm<128, 128, 32, 32, 0, 1, 1, 1><<<dim3( 256, 8), blk, 0, stream>>>(
      t1b, pC1, b_conv1, out1,    o1b, zbuf);
  dec_gemm<128,  64, 32, 64, 1, 0, 0, 1><<<dim3(2048, 4), blk, 0, stream>>>(
      o1b, pU2, b_up2,   nullptr, t2b, zbuf);
  dec_gemm< 64,   3, 64, 64, 0, 0, 1, 0><<<dim3(2048, 1), blk, 0, stream>>>(
      t2b, pC2, b_conv2, out0,    nullptr, zbuf);
}
